// RoadAttention_62088047231244
// MI455X (gfx1250) — matrix operand order repaired and checked by reading, not compiled, not run on hardware
//
#include <hip/hip_runtime.h>
#include <cstdint>

typedef __attribute__((ext_vector_type(16))) _Float16 v16h;
typedef __attribute__((ext_vector_type(8)))  _Float16 v8h;
typedef __attribute__((ext_vector_type(8)))  float    v8f;

constexpr int CH   = 128;     // hidden dim
constexpr int NMAP = 10000;
constexpr int NDST = 11520;
constexpr int DEG  = 32;
constexpr int ALD  = 136;     // A-tile LDS stride (halves): 272B, 16B-aligned rows
constexpr int CLD  = 132;     // C-tile LDS stride (floats)
constexpr float LN_EPS = 1e-5f;

// ---------------------------------------------------------------------------
// k-permutation so each lane's A fragment is 16 contiguous halves.
// Required wave32 A layout (ISA 7.12.2): lane<16 slots = K{0..7,16..23},
// lane>=16 slots = K{8..15,24..31}. Store feature k at position:
//   pos(k) = k ^ ( ((k>>3)^(k>>4))&1 ? 24 : 0 )   within each 32-block.
// ---------------------------------------------------------------------------
__device__ __host__ inline int aperm(int c) {
  int k = c & 31;
  int sw = (((k >> 3) ^ (k >> 4)) & 1) * 24;
  return (c & ~31) | (k ^ sw);
}

// A fragment: two contiguous 16B LDS loads, no repacking.
__device__ inline v16h frag_a(const _Float16* sA, int mt, int kb, int lane) {
  const _Float16* p =
      sA + (mt * 16 + (lane & 15)) * ALD + kb * 32 + (lane >> 4) * 16;
  v8h lo = *(const v8h*)p;
  v8h hi = *(const v8h*)(p + 8);
  return __builtin_shufflevector(lo, hi, 0, 1, 2, 3, 4, 5, 6, 7,
                                 8, 9, 10, 11, 12, 13, 14, 15);
}

// B fragment: weights pre-swizzled in global memory to [kb][nt][lane][16],
// so each lane reads 32 contiguous bytes (L2-resident).
__device__ inline v16h frag_b(const _Float16* __restrict__ Wf, int kb, int nt,
                              int lane) {
  return *(const v16h*)(Wf + (((kb * 8 + nt) * 32) + lane) * 16);
}

// 32x128 @ 128x128 GEMM: each wave accumulates 4 16x16 tiles. No barriers.
__device__ inline void gemm128(const _Float16* sA,
                               const _Float16* __restrict__ Wf, v8f acc[4],
                               int tid) {
  int lane = tid & 31;
  int mt = (tid >> 6) & 1;
  int ntb = ((tid >> 5) & 1) * 4;
#pragma unroll
  for (int kb = 0; kb < 4; ++kb) {
    v16h a = frag_a(sA, mt, kb, lane);
#pragma unroll
    for (int jj = 0; jj < 4; ++jj) {
      v16h b = frag_b(Wf, kb, ntb + jj, lane);
      acc[jj] = __builtin_amdgcn_wmma_f32_16x16x32_f16(
          false, a, false, b, (short)0, acc[jj], false, false);
    }
  }
}

#define ZERO4(A)                                                        \
  do {                                                                  \
    _Pragma("unroll") for (int jj = 0; jj < 4; ++jj) {                  \
      _Pragma("unroll") for (int q = 0; q < 8; ++q) (A)[jj][q] = 0.f;   \
    }                                                                   \
  } while (0)

// Row-wise LayerNorm of sC[32][128] -> f16 (k-permuted) into sA
__device__ inline void ln_rows(const float* sC, _Float16* sA,
                               const float* __restrict__ g,
                               const float* __restrict__ b, bool relu,
                               float* sM, float* sR, int tid) {
  if (tid < 32) {
    const float* row = sC + tid * CLD;
    float m = 0.f;
    for (int c = 0; c < CH; ++c) m += row[c];
    m *= (1.f / CH);
    float v = 0.f;
    for (int c = 0; c < CH; ++c) { float d = row[c] - m; v += d * d; }
    v *= (1.f / CH);
    sM[tid] = m;
    sR[tid] = rsqrtf(v + LN_EPS);
  }
  __syncthreads();
  int e = tid >> 2, cb = (tid & 3) * 32;
  float m = sM[e], r = sR[e];
  for (int j = 0; j < 32; ++j) {
    int c = cb + j;
    float val = (sC[e * CLD + c] - m) * r * g[c] + b[c];
    if (relu) val = fmaxf(val, 0.f);
    sA[e * ALD + aperm(c)] = (_Float16)val;
  }
  __syncthreads();
}

// ---------------------------------------------------------------------------
// Kernel 0: f32 -> f16 weight conversion + swizzle to fragment layout
//   fragment offset off = ((kb*8+nt)*32+lane)*16+j  maps to
//   row-major element [kb*32 + (lane>>4)*8 + (j&7) + (j>>3)*16][nt*16 + (lane&15)]
// ---------------------------------------------------------------------------
__device__ inline int frag_src(int off) {
  int kb = off >> 12;
  int nt = (off >> 9) & 7;
  int lane = (off >> 4) & 31;
  int j = off & 15;
  int col = nt * 16 + (lane & 15);
  int krow = kb * 32 + ((lane >> 4) << 3) + (j & 7) + ((j >> 3) << 4);
  return krow * 128 + col;
}

__global__ void wconv(const float* __restrict__ fW1, const float* __restrict__ fW2,
                      const float* __restrict__ fWo, const float* __restrict__ fWkr,
                      const float* __restrict__ fWvr,
                      _Float16* W1h, _Float16* W2h, _Float16* Woh,
                      _Float16* Wkrh, _Float16* Wvrh) {
  int idx = blockIdx.x * blockDim.x + threadIdx.x;
  if (idx < 3 * 128 * 128) {
    int i = idx >> 14, off = idx & 16383;
    int s = frag_src(off);
    W1h[idx] = (_Float16)fW1[i * 129 * 128 + s];  // rows 0..127 only
    W2h[idx] = (_Float16)fW2[i * 16384 + s];
  }
  if (idx < 128 * 128) {
    int s = frag_src(idx);
    Woh[idx]  = (_Float16)fWo[s];
    Wkrh[idx] = (_Float16)fWkr[s];
    Wvrh[idx] = (_Float16)fWvr[s];
  }
}

// ---------------------------------------------------------------------------
// Kernel 1: k/v projections of prenormed map nodes
// ---------------------------------------------------------------------------
__global__ __launch_bounds__(128) void kvproj(
    const float* __restrict__ mapx, const float* __restrict__ g,
    const float* __restrict__ b, const float* __restrict__ Wk,
    const float* __restrict__ Wv, const float* __restrict__ bv,
    float* __restrict__ kmap, float* __restrict__ vmap) {
  int row = blockIdx.x, tid = threadIdx.x;
  __shared__ float s1[128], s2[128], sx[128];
  float v = mapx[(size_t)row * CH + tid];
  s1[tid] = v; s2[tid] = v * v;
  __syncthreads();
  for (int off = 64; off; off >>= 1) {
    if (tid < off) { s1[tid] += s1[tid + off]; s2[tid] += s2[tid + off]; }
    __syncthreads();
  }
  float m = s1[0] * (1.f / CH);
  float var = s2[0] * (1.f / CH) - m * m;
  float xs = (v - m) * rsqrtf(var + LN_EPS) * g[tid] + b[tid];
  sx[tid] = xs;
  __syncthreads();
  float kk = 0.f, vv = bv[tid];
  for (int j = 0; j < CH; ++j) {
    float xj = sx[j];
    kk += xj * Wk[j * CH + tid];
    vv += xj * Wv[j * CH + tid];
  }
  kmap[(size_t)row * CH + tid] = kk;
  vmap[(size_t)row * CH + tid] = vv;
}

// ---------------------------------------------------------------------------
// Kernel 2: q projection + prenormed xd for dst nodes
// ---------------------------------------------------------------------------
__global__ __launch_bounds__(128) void dstproj(
    const float* __restrict__ dstx, const float* __restrict__ g,
    const float* __restrict__ b, const float* __restrict__ Wq,
    const float* __restrict__ bq, float* __restrict__ qg,
    float* __restrict__ xdg) {
  int row = blockIdx.x, tid = threadIdx.x;
  __shared__ float s1[128], s2[128], sx[128];
  float v = dstx[(size_t)row * CH + tid];
  s1[tid] = v; s2[tid] = v * v;
  __syncthreads();
  for (int off = 64; off; off >>= 1) {
    if (tid < off) { s1[tid] += s1[tid + off]; s2[tid] += s2[tid + off]; }
    __syncthreads();
  }
  float m = s1[0] * (1.f / CH);
  float var = s2[0] * (1.f / CH) - m * m;
  float xd = (v - m) * rsqrtf(var + LN_EPS) * g[tid] + b[tid];
  xdg[(size_t)row * CH + tid] = xd;
  sx[tid] = xd;
  __syncthreads();
  float q = bq[tid];
  for (int j = 0; j < CH; ++j) q += sx[j] * Wq[j * CH + tid];
  qg[(size_t)row * CH + tid] = q;
}

// ---------------------------------------------------------------------------
// Kernel 3: per-dst-node block: Fourier MLP + kr/vr + attention (WMMA)
// ---------------------------------------------------------------------------
struct CParams {
  const int* src;
  const float* pl_rel;
  const float* freqs;
  const float* fW1;   const float* fb1;
  const float* fln1g; const float* fln1b;
  const float* fb2;
  const float* flnog; const float* flnob;
  const float* fbo;
  const float* pnrg;  const float* pnrb;
  const float* bvr;
  const float* kmap;  const float* vmap; const float* qg;
  float* aggg;
  const _Float16* W1h; const _Float16* W2h; const _Float16* Woh;
  const _Float16* Wkrh; const _Float16* Wvrh;
};

__global__ __launch_bounds__(128) void road_edge(CParams P) {
  __shared__ __align__(16) _Float16 sA[32 * ALD];
  __shared__ float sC[32 * CLD];
  __shared__ float sM[32], sR[32];
  __shared__ float sQ[128], sB2[128], sXval[32], sSim[256], sAtt[256];
  __shared__ int sSrc[32];

  int n = blockIdx.x, tid = threadIdx.x, lane = tid & 31;
  int mt = (tid >> 6) & 1, ntb = ((tid >> 5) & 1) * 4;
  int col = lane & 15, rb0 = mt * 16 + 8 * (lane >> 4);

  if (tid < 32) sSrc[tid] = P.src[n * DEG + tid];
  sQ[tid]  = P.qg[(size_t)n * CH + tid];
  sB2[tid] = P.fb2[tid] + P.fb2[128 + tid] + P.fb2[256 + tid];

  v8f accS[4];
  ZERO4(accS);

  for (int i = 0; i < 3; ++i) {
    __syncthreads();  // prior readers of sA are done
    {   // Fourier features for input dim i -> sA[32][128] f16 (k-permuted)
      int e = tid & 31, p = tid >> 5;
      float x = P.pl_rel[(size_t)(n * DEG + e) * 3 + i];
      if (p == 0) sXval[e] = x;
      const float* fq = P.freqs + i * 64;
#pragma unroll 4
      for (int j = 0; j < 16; ++j) {
        int f = p * 16 + j;
        float ph = x * fq[f] * 6.28318530717958647692f;
        sA[e * ALD + aperm(f)]      = (_Float16)cosf(ph);
        sA[e * ALD + aperm(64 + f)] = (_Float16)sinf(ph);
      }
    }
    __syncthreads();  // features visible before fragment loads
    // GEMM1: feat[32,128] @ W1_i[128,128]  (+ rank-1 x*W1[128,:] + b1)
    v8f acc1[4];
    ZERO4(acc1);
    gemm128(sA, P.W1h + i * 16384, acc1, tid);
#pragma unroll
    for (int jj = 0; jj < 4; ++jj) {
      int c = (ntb + jj) * 16 + col;
      float w1l = P.fW1[i * 129 * 128 + 128 * 128 + c];
      float bb  = P.fb1[i * 128 + c];
#pragma unroll
      for (int v = 0; v < 8; ++v) {
        int e = rb0 + v;
        sC[e * CLD + c] = acc1[jj][v] + sXval[e] * w1l + bb;
      }
    }
    __syncthreads();
    ln_rows(sC, sA, P.fln1g + i * 128, P.fln1b + i * 128, true, sM, sR, tid);
    // GEMM2: a1 @ W2_i, accumulated over the 3 input dims
    gemm128(sA, P.W2h + i * 16384, accS, tid);
  }
  __syncthreads();
#pragma unroll
  for (int jj = 0; jj < 4; ++jj) {           // s = sum_i h_i + sum_i b2_i
    int c = (ntb + jj) * 16 + col;
    float bb = sB2[c];
#pragma unroll
    for (int v = 0; v < 8; ++v) sC[(rb0 + v) * CLD + c] = accS[jj][v] + bb;
  }
  __syncthreads();
  ln_rows(sC, sA, P.flnog, P.flnob, true, sM, sR, tid);

  // GEMM3: r = a_s @ f_Wo + f_bo
  v8f acc3[4];
  ZERO4(acc3);
  gemm128(sA, P.Woh, acc3, tid);
#pragma unroll
  for (int jj = 0; jj < 4; ++jj) {
    int c = (ntb + jj) * 16 + col;
    float bb = P.fbo[c];
#pragma unroll
    for (int v = 0; v < 8; ++v) sC[(rb0 + v) * CLD + c] = acc3[jj][v] + bb;
  }
  __syncthreads();
  ln_rows(sC, sA, P.pnrg, P.pnrb, false, sM, sR, tid);  // rn

  // GEMM4: kr = rn @ Wkr ; k_j = k[src] + kr
  v8f acc4[4];
  ZERO4(acc4);
  gemm128(sA, P.Wkrh, acc4, tid);
#pragma unroll
  for (int jj = 0; jj < 4; ++jj) {
    int c = (ntb + jj) * 16 + col;
#pragma unroll
    for (int v = 0; v < 8; ++v) {
      int e = rb0 + v;
      sC[e * CLD + c] = acc4[jj][v] + P.kmap[(size_t)sSrc[e] * CH + c];
    }
  }
  __syncthreads();
  // sim[e,h] = <q_h, k_j[e,h]> * HEAD_DIM^-0.5
#pragma unroll
  for (int p2 = 0; p2 < 2; ++p2) {
    int pair = tid + p2 * 128;
    int e2 = pair >> 3, h2 = pair & 7;
    float s = 0.f;
#pragma unroll
    for (int d = 0; d < 16; ++d) s += sQ[h2 * 16 + d] * sC[e2 * CLD + h2 * 16 + d];
    sSim[e2 * 8 + h2] = s * 0.25f;
  }
  __syncthreads();  // sim reads of sC complete before GEMM5 writeback
  // GEMM5: vr = rn @ Wvr + bvr ; v_j = v[src] + vr
  v8f acc5[4];
  ZERO4(acc5);
  gemm128(sA, P.Wvrh, acc5, tid);
#pragma unroll
  for (int jj = 0; jj < 4; ++jj) {
    int c = (ntb + jj) * 16 + col;
    float bb = P.bvr[c];
#pragma unroll
    for (int v = 0; v < 8; ++v) {
      int e = rb0 + v;
      sC[e * CLD + c] = acc5[jj][v] + P.vmap[(size_t)sSrc[e] * CH + c] + bb;
    }
  }
  __syncthreads();
  // segment softmax over the 32 contiguous edges of this dst, per head
  if (tid < 8) {
    int h2 = tid;
    float mx = -3.4e38f;
    for (int e = 0; e < 32; ++e) mx = fmaxf(mx, sSim[e * 8 + h2]);
    float den = 1e-16f;
    for (int e = 0; e < 32; ++e) {
      float ex = expf(sSim[e * 8 + h2] - mx);
      sAtt[e * 8 + h2] = ex;
      den += ex;
    }
    float inv = 1.f / den;
    for (int e = 0; e < 32; ++e) sAtt[e * 8 + h2] *= inv;
  }
  __syncthreads();
  {   // agg[c] = sum_e attn[e, c/16] * v_j[e, c]
    int h2 = tid >> 4;
    float a = 0.f;
#pragma unroll 4
    for (int e = 0; e < 32; ++e) a += sAtt[e * 8 + h2] * sC[e * CLD + tid];
    P.aggg[(size_t)n * CH + tid] = a;
  }
}

// ---------------------------------------------------------------------------
// Kernel 4: gated update + output proj + FFN + residuals
// ---------------------------------------------------------------------------
struct DParams {
  const float* dstx;
  const float* aggg; const float* xdg;
  const float* Wg; const float* bg;
  const float* Ws; const float* bs;
  const float* Wo; const float* bo;
  const float* postng; const float* postnb;
  const float* ffpng;  const float* ffpnb;
  const float* ffW1;   const float* ffb1;
  const float* ffW2;   const float* ffb2;
  const float* ffpostg; const float* ffpostb;
  float* out;
};

__device__ inline float block_ln(float v, float g, float b, float* s1, float* s2,
                                 int tid) {
  __syncthreads();
  s1[tid] = v; s2[tid] = v * v;
  __syncthreads();
  for (int off = 64; off; off >>= 1) {
    if (tid < off) { s1[tid] += s1[tid + off]; s2[tid] += s2[tid + off]; }
    __syncthreads();
  }
  float m = s1[0] * (1.f / 128.f);
  float var = s2[0] * (1.f / 128.f) - m * m;
  return (v - m) * rsqrtf(var + LN_EPS) * g + b;
}

__global__ __launch_bounds__(128) void fuse_out(DParams P) {
  int n = blockIdx.x, tid = threadIdx.x;
  __shared__ float sAgg[128], sXd[128], sTmp[128], sH1[512], s1[128], s2[128];
  sAgg[tid] = P.aggg[(size_t)n * CH + tid];
  sXd[tid]  = P.xdg[(size_t)n * CH + tid];
  __syncthreads();
  float gs = P.bg[tid];
  for (int j = 0; j < CH; ++j) gs += sAgg[j] * P.Wg[j * CH + tid];
  for (int j = 0; j < CH; ++j) gs += sXd[j] * P.Wg[(CH + j) * CH + tid];
  float gv = 1.f / (1.f + expf(-gs));
  float ss = P.bs[tid];
  for (int j = 0; j < CH; ++j) ss += sXd[j] * P.Ws[j * CH + tid];
  float a2 = sAgg[tid] + gv * (ss - sAgg[tid]);
  __syncthreads();
  sTmp[tid] = a2;
  __syncthreads();
  float ao = P.bo[tid];
  for (int j = 0; j < CH; ++j) ao += sTmp[j] * P.Wo[j * CH + tid];
  float aoln = block_ln(ao, P.postng[tid], P.postnb[tid], s1, s2, tid);
  float x = P.dstx[(size_t)n * CH + tid] + aoln;
  float hn = block_ln(x, P.ffpng[tid], P.ffpnb[tid], s1, s2, tid);
  __syncthreads();
  sTmp[tid] = hn;
  __syncthreads();
#pragma unroll
  for (int u = 0; u < 4; ++u) {
    int i1 = u * 128 + tid;
    float h1 = P.ffb1[i1];
    for (int j = 0; j < CH; ++j) h1 += sTmp[j] * P.ffW1[j * 512 + i1];
    sH1[i1] = fmaxf(h1, 0.f);
  }
  __syncthreads();
  float f = P.ffb2[tid];
  for (int i = 0; i < 512; ++i) f += sH1[i] * P.ffW2[i * CH + tid];
  float fln = block_ln(f, P.ffpostg[tid], P.ffpostb[tid], s1, s2, tid);
  P.out[(size_t)n * CH + tid] = x + fln;
}

// ---------------------------------------------------------------------------
extern "C" void kernel_launch(void* const* d_in, const int* in_sizes, int n_in,
                              void* d_out, int out_size, void* d_ws,
                              size_t ws_size, hipStream_t stream) {
  (void)in_sizes; (void)n_in; (void)out_size; (void)ws_size;
  const float* map_x  = (const float*)d_in[0];
  const int*   edges  = (const int*)d_in[1];     // [2, E], row0 = src
  const float* pl_rel = (const float*)d_in[2];
  const float* dst_x  = (const float*)d_in[3];
  const float* freqs  = (const float*)d_in[4];
  const float* fW1    = (const float*)d_in[5];
  const float* fb1    = (const float*)d_in[6];
  const float* fln1g  = (const float*)d_in[7];
  const float* fln1b  = (const float*)d_in[8];
  const float* fW2    = (const float*)d_in[9];
  const float* fb2    = (const float*)d_in[10];
  const float* flnog  = (const float*)d_in[11];
  const float* flnob  = (const float*)d_in[12];
  const float* fWo    = (const float*)d_in[13];
  const float* fbo    = (const float*)d_in[14];
  const float* pnsrcg = (const float*)d_in[15];
  const float* pnsrcb = (const float*)d_in[16];
  const float* pndstg = (const float*)d_in[17];
  const float* pndstb = (const float*)d_in[18];
  const float* pnrg   = (const float*)d_in[19];
  const float* pnrb   = (const float*)d_in[20];
  const float* postng = (const float*)d_in[21];
  const float* postnb = (const float*)d_in[22];
  const float* ffpng  = (const float*)d_in[23];
  const float* ffpnb  = (const float*)d_in[24];
  const float* ffpostg= (const float*)d_in[25];
  const float* ffpostb= (const float*)d_in[26];
  const float* Wq     = (const float*)d_in[27];
  const float* bq     = (const float*)d_in[28];
  const float* Wk     = (const float*)d_in[29];
  const float* Wv     = (const float*)d_in[30];
  const float* bv     = (const float*)d_in[31];
  const float* Wkr    = (const float*)d_in[32];
  const float* Wvr    = (const float*)d_in[33];
  const float* bvr    = (const float*)d_in[34];
  const float* Ws     = (const float*)d_in[35];
  const float* bs     = (const float*)d_in[36];
  const float* Wg     = (const float*)d_in[37];
  const float* bg     = (const float*)d_in[38];
  const float* Wo     = (const float*)d_in[39];
  const float* bo     = (const float*)d_in[40];
  const float* ffW1   = (const float*)d_in[41];
  const float* ffb1   = (const float*)d_in[42];
  const float* ffW2   = (const float*)d_in[43];
  const float* ffb2   = (const float*)d_in[44];

  float* ws = (float*)d_ws;
  size_t o = 0;
  float* kmap = ws + o; o += (size_t)NMAP * CH;
  float* vmap = ws + o; o += (size_t)NMAP * CH;
  float* qg   = ws + o; o += (size_t)NDST * CH;
  float* xdg  = ws + o; o += (size_t)NDST * CH;
  float* aggg = ws + o; o += (size_t)NDST * CH;
  _Float16* W1h  = (_Float16*)(ws + o);
  _Float16* W2h  = W1h + 3 * 16384;
  _Float16* Woh  = W2h + 3 * 16384;
  _Float16* Wkrh = Woh + 16384;
  _Float16* Wvrh = Wkrh + 16384;

  wconv<<<192, 256, 0, stream>>>(fW1, fW2, fWo, Wkr, Wvr, W1h, W2h, Woh, Wkrh, Wvrh);
  kvproj<<<NMAP, 128, 0, stream>>>(map_x, pnsrcg, pnsrcb, Wk, Wv, bv, kmap, vmap);
  dstproj<<<NDST, 128, 0, stream>>>(dst_x, pndstg, pndstb, Wq, bq, qg, xdg);

  CParams cp;
  cp.src = edges;  // first N_EDGES entries = src row
  cp.pl_rel = pl_rel; cp.freqs = freqs;
  cp.fW1 = fW1; cp.fb1 = fb1; cp.fln1g = fln1g; cp.fln1b = fln1b;
  cp.fb2 = fb2; cp.flnog = flnog; cp.flnob = flnob; cp.fbo = fbo;
  cp.pnrg = pnrg; cp.pnrb = pnrb; cp.bvr = bvr;
  cp.kmap = kmap; cp.vmap = vmap; cp.qg = qg; cp.aggg = aggg;
  cp.W1h = W1h; cp.W2h = W2h; cp.Woh = Woh; cp.Wkrh = Wkrh; cp.Wvrh = Wvrh;
  road_edge<<<NDST, 128, 0, stream>>>(cp);

  DParams dp;
  dp.dstx = dst_x; dp.aggg = aggg; dp.xdg = xdg;
  dp.Wg = Wg; dp.bg = bg; dp.Ws = Ws; dp.bs = bs; dp.Wo = Wo; dp.bo = bo;
  dp.postng = postng; dp.postnb = postnb;
  dp.ffpng = ffpng; dp.ffpnb = ffpnb;
  dp.ffW1 = ffW1; dp.ffb1 = ffb1; dp.ffW2 = ffW2; dp.ffb2 = ffb2;
  dp.ffpostg = ffpostg; dp.ffpostb = ffpostb;
  dp.out = (float*)d_out;
  fuse_out<<<NDST, 128, 0, stream>>>(dp);
}
